// AttentionBlock_13915694039828
// MI455X (gfx1250) — compile-verified, matrix operand order checked by reading, not run on hardware
//
#include <hip/hip_runtime.h>
#include <hip/hip_bf16.h>

typedef __attribute__((ext_vector_type(16))) _Float16 v16h;
typedef __attribute__((ext_vector_type(8)))  _Float16 v8h;
typedef __attribute__((ext_vector_type(8)))  float    v8f;

#define F_DIM  2048
#define BOTTLE 512
#define BATCH  64
#define EPS    1e-5f
#define SLOPE  0.01f

// ---------------- plain f32 -> f16 conversion (activations) ----------------
__global__ void f32_to_f16(const float* __restrict__ in, _Float16* __restrict__ out, int n) {
  int i = blockIdx.x * blockDim.x + threadIdx.x;
  if (i < n) out[i] = (_Float16)in[i];
}

// ---------------- tiled transpose + f32->f16: Wt[n*rows + k] = W[k*cols + n] ----------
__global__ void transpose_to_f16(const float* __restrict__ in, _Float16* __restrict__ out,
                                 int rows, int cols) {
  __shared__ float tile[32][33];
  const int c = blockIdx.x * 32 + threadIdx.x;
  const int r = blockIdx.y * 32 + threadIdx.y;
#pragma unroll
  for (int j = 0; j < 32; j += 8)
    if (r + j < rows && c < cols) tile[threadIdx.y + j][threadIdx.x] = in[(size_t)(r + j) * cols + c];
  __syncthreads();
  const int tc = blockIdx.y * 32 + threadIdx.x;  // out col = in row
  const int tr = blockIdx.x * 32 + threadIdx.y;  // out row = in col
#pragma unroll
  for (int j = 0; j < 32; j += 8)
    if (tr + j < cols && tc < rows)
      out[(size_t)(tr + j) * rows + tc] = (_Float16)tile[threadIdx.x][threadIdx.y + j];
}

// ---------------- WMMA GEMM: C[M,N] = act(A[M,K] @ Bt[N,K]^T + bias) ----------------
// Bt pre-transposed to [N,K] -> A and B fragments are each two contiguous 16-byte
// loads per lane. One wave32 computes a 16x64 slab (4 N-tiles). All 4 B fragments
// are loaded into distinct registers BEFORE the 4 back-to-back WMMAs so the loads
// of one k-step stay in flight together instead of draining before every matrix op.
template <int M, int N, int K, int LEAKY, int F16OUT>
__global__ void gemm_wmma(const _Float16* __restrict__ A, const _Float16* __restrict__ Bt,
                          const float* __restrict__ bias, void* __restrict__ Cout) {
  constexpr int NT = 4;                  // N-tiles per wave
  constexpr int WN = N / (16 * NT);      // waves along N
  const int lane = threadIdx.x & 31;
  const int wave = (blockIdx.x * blockDim.x + threadIdx.x) >> 5;
  if (wave >= (M / 16) * WN) return;
  const int tm = wave / WN;
  const int n0 = (wave % WN) * (16 * NT);

  const int row = tm * 16 + (lane & 15);
  const int kb  = (lane >> 4) * 8;       // lanes 16-31 supply K+8..15 / K+24..31
  const _Float16* __restrict__ arow = A + (size_t)row * K + kb;
  const _Float16* __restrict__ bbase = Bt + (size_t)(n0 + (lane & 15)) * K + kb;

  v8f c[NT] = {};
  union Pack { v16h v; v8h h[2]; };
#pragma unroll 2
  for (int k0 = 0; k0 < K; k0 += 32) {
    Pack a, b[NT];
    a.h[0] = *(const v8h*)(arow + k0);
    a.h[1] = *(const v8h*)(arow + k0 + 16);
#pragma unroll
    for (int t = 0; t < NT; ++t) {
      b[t].h[0] = *(const v8h*)(bbase + (size_t)t * 16 * K + k0);
      b[t].h[1] = *(const v8h*)(bbase + (size_t)t * 16 * K + k0 + 16);
    }
    __builtin_prefetch(arow + k0 + 32, 0, 3);
#pragma unroll
    for (int t = 0; t < NT; ++t) {
      __builtin_prefetch(bbase + (size_t)t * 16 * K + k0 + 32, 0, 3);
      c[t] = __builtin_amdgcn_wmma_f32_16x16x32_f16(false, a.v, false, b[t].v,
                                                    (short)0, c[t], false, false);
    }
  }

  // C/D layout: VGPR r -> row (r + 8*(lane>=16)), col lane&15 within each 16x16 tile.
  const int rbase = (lane >> 4) * 8;
#pragma unroll
  for (int t = 0; t < NT; ++t) {
    const int   col = n0 + t * 16 + (lane & 15);
    const float bv  = bias[col];
#pragma unroll
    for (int r = 0; r < 8; ++r) {
      const int rowo = tm * 16 + rbase + r;
      float v = c[t][r] + bv;
      if (LEAKY) v = (v >= 0.f) ? v : SLOPE * v;
      if (F16OUT) ((_Float16*)Cout)[(size_t)rowo * N + col] = (_Float16)v;
      else        ((float*)Cout)[(size_t)rowo * N + col] = v;
    }
  }
}

// ---------------- per-batch max/min of q (for stable softmax) ----------------
__global__ void q_minmax(const float* __restrict__ q, float* __restrict__ qmx,
                         float* __restrict__ qmn) {
  __shared__ float smx[256], smn[256];
  const int b = blockIdx.x;
  float mx = -1e30f, mn = 1e30f;
  for (int i = threadIdx.x; i < F_DIM; i += 256) {
    const float v = q[b * F_DIM + i];
    mx = fmaxf(mx, v); mn = fminf(mn, v);
  }
  smx[threadIdx.x] = mx; smn[threadIdx.x] = mn;
  __syncthreads();
  for (int s = 128; s > 0; s >>= 1) {
    if (threadIdx.x < s) {
      smx[threadIdx.x] = fmaxf(smx[threadIdx.x], smx[threadIdx.x + s]);
      smn[threadIdx.x] = fminf(smn[threadIdx.x], smn[threadIdx.x + s]);
    }
    __syncthreads();
  }
  if (threadIdx.x == 0) { qmx[b] = smx[0]; qmn[b] = smn[0]; }
}

// ---------------- fused rank-1 softmax-attention + residual ----------------
// scores[b,i,:] = softmax(k[b,i] * q[b,:]) -> never materialized; one wave32 per row.
__global__ void attn_rank1(const float* __restrict__ x, const float* __restrict__ q,
                           const float* __restrict__ k, const float* __restrict__ qmx,
                           const float* __restrict__ qmn, float* __restrict__ attn) {
  const int lane = threadIdx.x & 31;
  const int w    = (blockIdx.x * blockDim.x + threadIdx.x) >> 5;
  const int b    = w >> 11;            // / F_DIM
  const int i    = w & (F_DIM - 1);
  const float kv = k[b * F_DIM + i];
  const float m  = (kv >= 0.f) ? kv * qmx[b] : kv * qmn[b];
  const float* qb = q + b * F_DIM;
  const float* xb = x + b * F_DIM;
  float se = 0.f, sx = 0.f;
  for (int j = lane; j < F_DIM; j += 32) {
    const float e = __expf(kv * qb[j] - m);
    se += e;
    sx += e * xb[j];
  }
#pragma unroll
  for (int off = 16; off > 0; off >>= 1) {
    se += __shfl_xor(se, off, 32);
    sx += __shfl_xor(sx, off, 32);
  }
  if (lane == 0) attn[b * F_DIM + i] = sx / se + xb[i];
}

// ---------------- BatchNorm1d over batch axis (training stats, biased var) ----------------
__global__ void batchnorm(const float* __restrict__ a, const float* __restrict__ gamma,
                          const float* __restrict__ beta, float* __restrict__ out) {
  const int i = blockIdx.x * blockDim.x + threadIdx.x;
  if (i >= F_DIM) return;
  float s = 0.f, s2 = 0.f;
  for (int b = 0; b < BATCH; ++b) {
    const float v = a[b * F_DIM + i];
    s += v; s2 += v * v;
  }
  const float mean = s * (1.f / BATCH);
  const float var  = s2 * (1.f / BATCH) - mean * mean;
  const float sc   = gamma[i] * rsqrtf(var + EPS);
  const float sh   = beta[i] - mean * sc;
  for (int b = 0; b < BATCH; ++b)
    out[b * F_DIM + i] = a[b * F_DIM + i] * sc + sh;
}

extern "C" void kernel_launch(void* const* d_in, const int* in_sizes, int n_in,
                              void* d_out, int out_size, void* d_ws, size_t ws_size,
                              hipStream_t stream) {
  const float* x    = (const float*)d_in[0];
  const float* q_w1 = (const float*)d_in[1];
  const float* q_b1 = (const float*)d_in[2];
  const float* q_w2 = (const float*)d_in[3];
  const float* q_b2 = (const float*)d_in[4];
  const float* k_w1 = (const float*)d_in[5];
  const float* k_b1 = (const float*)d_in[6];
  const float* k_w2 = (const float*)d_in[7];
  const float* k_b2 = (const float*)d_in[8];
  const float* gam  = (const float*)d_in[9];
  const float* bet  = (const float*)d_in[10];
  float* out = (float*)d_out;

  // ---- workspace carve (16B aligned) ----
  char* ws = (char*)d_ws;
  size_t off = 0;
  auto carve = [&](size_t bytes) { char* p = ws + off; off += (bytes + 15) & ~size_t(15); return p; };
  _Float16* xh    = (_Float16*)carve((size_t)BATCH * F_DIM * 2);
  _Float16* qw1t  = (_Float16*)carve((size_t)F_DIM * BOTTLE * 2);  // [BOTTLE, F_DIM]
  _Float16* kw1t  = (_Float16*)carve((size_t)F_DIM * BOTTLE * 2);
  _Float16* qw2t  = (_Float16*)carve((size_t)BOTTLE * F_DIM * 2);  // [F_DIM, BOTTLE]
  _Float16* kw2t  = (_Float16*)carve((size_t)BOTTLE * F_DIM * 2);
  _Float16* hq    = (_Float16*)carve((size_t)BATCH * BOTTLE * 2);
  _Float16* hk    = (_Float16*)carve((size_t)BATCH * BOTTLE * 2);
  float*    qf    = (float*)carve((size_t)BATCH * F_DIM * 4);
  float*    kf    = (float*)carve((size_t)BATCH * F_DIM * 4);
  float*    at    = (float*)carve((size_t)BATCH * F_DIM * 4);
  float*    qmx   = (float*)carve(BATCH * 4);
  float*    qmn   = (float*)carve(BATCH * 4);

  // ---- activations: f32 -> f16 ----
  const int NX = BATCH * F_DIM;
  f32_to_f16<<<(NX + 255) / 256, 256, 0, stream>>>(x, xh, NX);

  // ---- weights: transpose + f32 -> f16 ----
  {
    dim3 blk(32, 8);
    dim3 g1(BOTTLE / 32, F_DIM / 32);   // w1: rows=F_DIM, cols=BOTTLE
    dim3 g2(F_DIM / 32, BOTTLE / 32);   // w2: rows=BOTTLE, cols=F_DIM
    transpose_to_f16<<<g1, blk, 0, stream>>>(q_w1, qw1t, F_DIM, BOTTLE);
    transpose_to_f16<<<g1, blk, 0, stream>>>(k_w1, kw1t, F_DIM, BOTTLE);
    transpose_to_f16<<<g2, blk, 0, stream>>>(q_w2, qw2t, BOTTLE, F_DIM);
    transpose_to_f16<<<g2, blk, 0, stream>>>(k_w2, kw2t, BOTTLE, F_DIM);
  }

  // ---- MLP layer 1: [64,2048] @ [2048,512], leaky ReLU, f16 out ----
  {
    const int waves = (BATCH / 16) * (BOTTLE / 64);      // 32
    const int blocks = (waves * 32 + 127) / 128;
    gemm_wmma<BATCH, BOTTLE, F_DIM, 1, 1><<<blocks, 128, 0, stream>>>(xh, qw1t, q_b1, hq);
    gemm_wmma<BATCH, BOTTLE, F_DIM, 1, 1><<<blocks, 128, 0, stream>>>(xh, kw1t, k_b1, hk);
  }
  // ---- MLP layer 2: [64,512] @ [512,2048], linear, f32 out ----
  {
    const int waves = (BATCH / 16) * (F_DIM / 64);       // 128
    const int blocks = (waves * 32 + 127) / 128;
    gemm_wmma<BATCH, F_DIM, BOTTLE, 0, 0><<<blocks, 128, 0, stream>>>(hq, qw2t, q_b2, qf);
    gemm_wmma<BATCH, F_DIM, BOTTLE, 0, 0><<<blocks, 128, 0, stream>>>(hk, kw2t, k_b2, kf);
  }

  // ---- per-batch q max/min ----
  q_minmax<<<BATCH, 256, 0, stream>>>(qf, qmx, qmn);

  // ---- fused rank-1 softmax attention + residual: one wave per row ----
  {
    const int rows = BATCH * F_DIM;                      // 131072 waves
    const int blocks = (rows * 32) / 256;                // 16384 blocks of 256
    attn_rank1<<<blocks, 256, 0, stream>>>(x, qf, kf, qmx, qmn, at);
  }

  // ---- batch norm -> d_out ----
  batchnorm<<<(F_DIM + 255) / 256, 256, 0, stream>>>(at, gam, bet, out);
}
